// MultiHeadAttention_17351667875848
// MI455X (gfx1250) — compile-verified, matrix operand order checked by reading, not run on hardware
//
#include <hip/hip_runtime.h>
#include <hip/hip_bf16.h>
#include <math.h>

typedef __attribute__((ext_vector_type(16))) _Float16 v16h;
typedef __attribute__((ext_vector_type(8)))  _Float16 v8h;
typedef __attribute__((ext_vector_type(8)))  float    v8f;
typedef __attribute__((ext_vector_type(4)))  float    v4f;
typedef __attribute__((ext_vector_type(4)))  int      v4i;

#define B_  2
#define S_  2048
#define D_  768
#define H_  12
#define DK_ 64

// CDNA5 async global->LDS DMA path (ASYNCcnt-tracked), if the toolchain
// exposes it; otherwise fall back to load+store staging.
#if __has_builtin(__builtin_amdgcn_global_load_async_to_lds_b128) && \
    __has_builtin(__builtin_amdgcn_s_wait_asynccnt)
#define HAVE_ASYNC_LDS 1
#else
#define HAVE_ASYNC_LDS 0
#endif

static __device__ __forceinline__ v8f wmma_f16(v16h a, v16h b, v8f c) {
  // D = A(16x32 f16) * B(32x16 f16) + C(16x16 f32)
  return __builtin_amdgcn_wmma_f32_16x16x32_f16(false, a, false, b, (short)0, c,
                                                false, false);
}

#if HAVE_ASYNC_LDS
static __device__ __forceinline__ void async_copy_b128(const _Float16* gsrc,
                                                       _Float16* ldst) {
  v4i* g = (v4i*)const_cast<_Float16*>(gsrc);
  v4i* l = (v4i*)ldst;
  __builtin_amdgcn_global_load_async_to_lds_b128(
      (__attribute__((address_space(1))) v4i*)g,
      (__attribute__((address_space(3))) v4i*)l, 0, 0);
}
#endif

// A (or B) fragment from a row-major tile: rows = M (A) / N (B), contiguous K.
// Lane L<16: row L, K stripes {0..7, 16..23}; lane L>=16: row L-16, stripes
// {8..15, 24..31}. Each stripe is a contiguous 16-byte run -> two b128 loads.
static __device__ __forceinline__ v16h frag_rm(const _Float16* base, int ld, int lane) {
  const _Float16* p = base + (lane & 15) * ld + ((lane >> 4) << 3);
  v8h lo = *(const v8h*)p;
  v8h hi = *(const v8h*)(p + 16);
  return __builtin_shufflevector(lo, hi, 0, 1, 2, 3, 4, 5, 6, 7,
                                 8, 9, 10, 11, 12, 13, 14, 15);
}

// ---------------------------------------------------------------------------
// C[m,n] = sum_d A[m,d] * W[n,d]  (y = x @ W^T), M = B*S = 4096, N = 768.
// IN_F32:  A is fp32 global input (converted to f16 while staging to LDS),
//          else A is f16 workspace (staged via async DMA when available).
// OUT_F32: write fp32 row-major [M,768] + bias (final projection),
//          else write f16 head-major [B,H,S,DK] (Q/K/V projections).
// Block: 128 threads (4 waves, 2x2), 64x64 output tile, K-step 32.
// ---------------------------------------------------------------------------
template <bool IN_F32, bool OUT_F32>
__global__ void __launch_bounds__(128)
gemm_xwT(const void* __restrict__ Ap, const float* __restrict__ W,
         const float* __restrict__ bias, void* __restrict__ Outp) {
  __shared__ alignas(16) _Float16 sA[64][40];   // 80 B row stride (16B mult)
  __shared__ alignas(16) _Float16 sB[64][40];
  const int tid  = threadIdx.x;
  const int wave = tid >> 5, lane = tid & 31;
  const int m0 = blockIdx.x * 64, n0 = blockIdx.y * 64;
  const int wm = (wave & 1) * 32, wn = (wave >> 1) * 32;

  v8f acc[2][2] = {};

  for (int k0 = 0; k0 < D_; k0 += 32) {
    // 64x32 halfs per tile = 256 chunks of 8; 128 threads x 2 chunks each.
#pragma unroll
    for (int i = 0; i < 2; ++i) {
      int chunk = tid + i * 128;
      int r = chunk >> 2, c = (chunk & 3) * 8;
      if (IN_F32) {
        v4f a0 = *(const v4f*)&((const float*)Ap)[(size_t)(m0 + r) * D_ + k0 + c];
        v4f a1 = *(const v4f*)&((const float*)Ap)[(size_t)(m0 + r) * D_ + k0 + c + 4];
        v8h av;
#pragma unroll
        for (int e = 0; e < 4; ++e) { av[e] = (_Float16)a0[e]; av[e + 4] = (_Float16)a1[e]; }
        *(v8h*)&sA[r][c] = av;
      } else {
        const _Float16* src = &((const _Float16*)Ap)[(size_t)(m0 + r) * D_ + k0 + c];
#if HAVE_ASYNC_LDS
        async_copy_b128(src, &sA[r][c]);      // DMA straight into LDS
#else
        *(v8h*)&sA[r][c] = *(const v8h*)src;
#endif
      }
      v4f b0 = *(const v4f*)&W[(size_t)(n0 + r) * D_ + k0 + c];
      v4f b1 = *(const v4f*)&W[(size_t)(n0 + r) * D_ + k0 + c + 4];
      v8h bv;
#pragma unroll
      for (int e = 0; e < 4; ++e) { bv[e] = (_Float16)b0[e]; bv[e + 4] = (_Float16)b1[e]; }
      *(v8h*)&sB[r][c] = bv;
    }
#if HAVE_ASYNC_LDS
    if (!IN_F32) __builtin_amdgcn_s_wait_asynccnt(0);
#endif
    __syncthreads();

    v16h a0 = frag_rm(&sA[wm][0],      40, lane);
    v16h a1 = frag_rm(&sA[wm + 16][0], 40, lane);
    v16h b0 = frag_rm(&sB[wn][0],      40, lane);
    v16h b1 = frag_rm(&sB[wn + 16][0], 40, lane);
    acc[0][0] = wmma_f16(a0, b0, acc[0][0]);
    acc[0][1] = wmma_f16(a0, b1, acc[0][1]);
    acc[1][0] = wmma_f16(a1, b0, acc[1][0]);
    acc[1][1] = wmma_f16(a1, b1, acc[1][1]);
    __syncthreads();
  }

  const int colL = lane & 15, rb = (lane >> 4) << 3;  // C frag: M = rb+r, N = colL
#pragma unroll
  for (int i = 0; i < 2; ++i)
#pragma unroll
    for (int j = 0; j < 2; ++j)
#pragma unroll
      for (int r = 0; r < 8; ++r) {
        int m = m0 + wm + i * 16 + rb + r;
        int n = n0 + wn + j * 16 + colL;
        float v = acc[i][j][r];
        if (OUT_F32) {
          ((float*)Outp)[(size_t)m * D_ + n] = v + bias[n];
        } else {
          int bb = m / S_, s = m - bb * S_;
          int h = n >> 6, dk = n & 63;
          ((_Float16*)Outp)[(((size_t)(bb * H_ + h) * S_ + s) << 6) + dk] =
              (_Float16)v;
        }
      }
}

// ---------------------------------------------------------------------------
// Flash attention forward. One block per (b, h, 64-query tile); 4 waves,
// each wave owns 16 query rows. K/V streamed in 32-key tiles through LDS.
// K tile staged via async DMA (when available); V stored transposed so its
// B-fragments are contiguous b128 reads.
// ---------------------------------------------------------------------------
__global__ void __launch_bounds__(128)
attn_fwd(const _Float16* __restrict__ Qh, const _Float16* __restrict__ Kh,
         const _Float16* __restrict__ Vh, _Float16* __restrict__ AttOut) {
  __shared__ alignas(16) _Float16 sK[32][72];    // [key][dk], 144 B stride
  __shared__ alignas(16) _Float16 sVt[64][40];   // [dk][key], transposed
  __shared__ alignas(16) float    sSc[4][16][32];
  __shared__ alignas(16) _Float16 sP[4][16][40];
  __shared__ float                sFac[4][16];

  const int tid = threadIdx.x, wave = tid >> 5, lane = tid & 31;
  const int nqb  = S_ / 64;
  const int qblk = blockIdx.x % nqb;
  const int bh   = blockIdx.x / nqb;
  const int b    = bh / H_, h = bh - b * H_;
  const _Float16* Q = Qh + (size_t)bh * S_ * DK_;
  const _Float16* K = Kh + (size_t)bh * S_ * DK_;
  const _Float16* V = Vh + (size_t)bh * S_ * DK_;

  const int q0   = qblk * 64 + wave * 16;    // wave's first query row
  const int colL = lane & 15, rb = (lane >> 4) << 3;

  // Resident Q A-fragments (dk chunks 0..31 and 32..63): four b128 loads.
  v16h qa0, qa1;
  {
    const _Float16* qp = Q + (size_t)(q0 + (lane & 15)) * DK_ + ((lane >> 4) << 3);
    v8h q00 = *(const v8h*)qp;
    v8h q01 = *(const v8h*)(qp + 16);
    v8h q10 = *(const v8h*)(qp + 32);
    v8h q11 = *(const v8h*)(qp + 48);
    qa0 = __builtin_shufflevector(q00, q01, 0, 1, 2, 3, 4, 5, 6, 7,
                                  8, 9, 10, 11, 12, 13, 14, 15);
    qa1 = __builtin_shufflevector(q10, q11, 0, 1, 2, 3, 4, 5, 6, 7,
                                  8, 9, 10, 11, 12, 13, 14, 15);
  }

  float m_i = -1e30f, l_i = 0.f;             // lanes 0..15: row = lane
  v8f o0 = {}, o1 = {}, o2 = {}, o3 = {};

  const int kend = qblk * 64 + 64;           // causal horizon for this block
  for (int kb0 = 0; kb0 < kend; kb0 += 32) {
    // K/V tiles: 32x64 halfs = 256 chunks of 8; 2 chunks per thread.
#pragma unroll
    for (int i = 0; i < 2; ++i) {
      int chunk = tid + i * 128;
      int r = chunk >> 3, c = (chunk & 7) * 8;          // key r, dk c..c+7
#if HAVE_ASYNC_LDS
      async_copy_b128(&K[(size_t)(kb0 + r) * DK_ + c], &sK[r][c]);
#else
      *(v8h*)&sK[r][c] = *(const v8h*)&K[(size_t)(kb0 + r) * DK_ + c];
#endif
      v8h vv = *(const v8h*)&V[(size_t)(kb0 + r) * DK_ + c];
#pragma unroll
      for (int j = 0; j < 8; ++j) sVt[c + j][r] = vv[j]; // transpose V
    }
    // prefetch next tile into L2 while we compute (global_prefetch_b8)
    __builtin_prefetch(&K[(size_t)(kb0 + 32) * DK_ + (tid << 4)], 0, 0);
    __builtin_prefetch(&V[(size_t)(kb0 + 32) * DK_ + (tid << 4)], 0, 0);
#if HAVE_ASYNC_LDS
    __builtin_amdgcn_s_wait_asynccnt(0);
#endif
    __syncthreads();

    // scores: S = Q(16x64) * K^T(64x32), as 2 N-tiles x 2 K-chunks
    v8f sc0 = {}, sc1 = {};
    sc0 = wmma_f16(qa0, frag_rm(&sK[0][0],   72, lane), sc0);
    sc0 = wmma_f16(qa1, frag_rm(&sK[0][32],  72, lane), sc0);
    sc1 = wmma_f16(qa0, frag_rm(&sK[16][0],  72, lane), sc1);
    sc1 = wmma_f16(qa1, frag_rm(&sK[16][32], 72, lane), sc1);

    // scale + causal mask -> LDS score tile (per wave, 16x32 fp32)
#pragma unroll
    for (int r = 0; r < 8; ++r) {
      int row  = rb + r;
      float v0 = sc0[r] * 0.125f;            // 1/sqrt(DK)
      float v1 = sc1[r] * 0.125f;
      if (kb0 + colL      > q0 + row) v0 = -1e30f;
      if (kb0 + 16 + colL > q0 + row) v1 = -1e30f;
      sSc[wave][row][colL]      = v0;
      sSc[wave][row][16 + colL] = v1;
    }
    __syncthreads();

    // online softmax: lane r < 16 owns row r of this wave's tile
    if (lane < 16) {
      float mx = m_i;
#pragma unroll
      for (int j = 0; j < 32; ++j) mx = fmaxf(mx, sSc[wave][lane][j]);
      float corr = __expf(m_i - mx);
      float sum = 0.f;
#pragma unroll
      for (int j = 0; j < 32; ++j) {
        float p = __expf(sSc[wave][lane][j] - mx);
        sP[wave][lane][j] = (_Float16)p;
        sum += p;
      }
      l_i = l_i * corr + sum;
      m_i = mx;
      sFac[wave][lane] = corr;
    }
    __syncthreads();

    // rescale O by row correction, then O += P(16x32) * V(32x64)
#pragma unroll
    for (int r = 0; r < 8; ++r) {
      float c = sFac[wave][rb + r];
      o0[r] *= c; o1[r] *= c; o2[r] *= c; o3[r] *= c;
    }
    v16h pa = frag_rm(&sP[wave][0][0], 40, lane);
    o0 = wmma_f16(pa, frag_rm(&sVt[0][0],  40, lane), o0);
    o1 = wmma_f16(pa, frag_rm(&sVt[16][0], 40, lane), o1);
    o2 = wmma_f16(pa, frag_rm(&sVt[32][0], 40, lane), o2);
    o3 = wmma_f16(pa, frag_rm(&sVt[48][0], 40, lane), o3);
    __syncthreads();
  }

  // finalize: divide by row sum, store attended as f16 [B,S,H*DK]
  if (lane < 16) sFac[wave][lane] = 1.0f / l_i;
  __syncthreads();
  _Float16* out = AttOut + ((size_t)b * S_ + q0) * D_ + h * DK_;
#pragma unroll
  for (int r = 0; r < 8; ++r) {
    float c = sFac[wave][rb + r];
    size_t ro = (size_t)(rb + r) * D_;
    out[ro + colL]      = (_Float16)(o0[r] * c);
    out[ro + 16 + colL] = (_Float16)(o1[r] * c);
    out[ro + 32 + colL] = (_Float16)(o2[r] * c);
    out[ro + 48 + colL] = (_Float16)(o3[r] * c);
  }
}

// ---------------------------------------------------------------------------
extern "C" void kernel_launch(void* const* d_in, const int* in_sizes, int n_in,
                              void* d_out, int out_size, void* d_ws,
                              size_t ws_size, hipStream_t stream) {
  (void)in_sizes; (void)n_in; (void)out_size; (void)ws_size;
  const float* query = (const float*)d_in[0];
  const float* key_  = (const float*)d_in[1];
  const float* value = (const float*)d_in[2];
  // d_in[3] is the causal mask; applied analytically in attn_fwd.
  const float* wq = (const float*)d_in[4];
  const float* wk = (const float*)d_in[5];
  const float* wv = (const float*)d_in[6];
  const float* wo = (const float*)d_in[7];
  const float* bo = (const float*)d_in[8];

  const size_t headElems = (size_t)B_ * H_ * S_ * DK_;  // 3,145,728
  _Float16* qh  = (_Float16*)d_ws;
  _Float16* kh  = qh + headElems;
  _Float16* vh  = kh + headElems;
  _Float16* att = vh + headElems;                       // B*S*D f16

  dim3 gProj(B_ * S_ / 64, D_ / 64);                    // 64 x 12 blocks
  gemm_xwT<true, false><<<gProj, 128, 0, stream>>>(query, wq, nullptr, qh);
  gemm_xwT<true, false><<<gProj, 128, 0, stream>>>(key_, wk, nullptr, kh);
  gemm_xwT<true, false><<<gProj, 128, 0, stream>>>(value, wv, nullptr, vh);

  attn_fwd<<<B_ * H_ * (S_ / 64), 128, 0, stream>>>(qh, kh, vh, att);

  gemm_xwT<false, true><<<gProj, 128, 0, stream>>>(att, wo, bo, d_out);
}
